// BloodSupplyGNN_47897475285043
// MI455X (gfx1250) — matrix operand order repaired and checked by reading, not compiled, not run on hardware
//
#include <hip/hip_runtime.h>
#include <hip/hip_bf16.h>

typedef __attribute__((ext_vector_type(16))) __bf16 v16bf;
typedef __attribute__((ext_vector_type(8)))  float  v8f;

#define HH   256
#define F_IN 32

// ---------------------------------------------------------------------------
// Fill helpers (graph-capture safe; we do not use hipMemsetAsync)
// ---------------------------------------------------------------------------
__global__ __launch_bounds__(256) void fill_f32(float* p, float v, long long n) {
    long long i = (long long)blockIdx.x * blockDim.x + threadIdx.x;
    if (i < n) p[i] = v;
}
__global__ __launch_bounds__(256) void fill_u32(unsigned* p, unsigned v, long long n) {
    long long i = (long long)blockIdx.x * blockDim.x + threadIdx.x;
    if (i < n) p[i] = v;
}

// ---------------------------------------------------------------------------
// Pack fp32 weight matrix W[K x N] (row-major) into WMMA bf16 B-fragments.
// Layout per (ktile, ntile): 32 lanes x 16 bf16 contiguous (32B per lane).
// B 32x16 in this code's contiguous k order: lanes 0-15 -> N=0..15, K=0..15 (elem i = K i);
//                       lanes 16-31 -> N=0..15, K=16..31.
// ---------------------------------------------------------------------------
__global__ __launch_bounds__(128)
void pack_wmma_b(const float* __restrict__ W, int K, int N, __bf16* __restrict__ out) {
    int t = blockIdx.x * blockDim.x + threadIdx.x;
    int ktiles = K >> 5, ntiles = N >> 4;
    int total = ktiles * ntiles * 32;
    if (t >= total) return;
    int lane = t & 31;
    int tile = t >> 5;
    int nt = tile % ntiles;
    int kt = tile / ntiles;
    int nn = nt * 16 + (lane & 15);
    int kbase = kt * 32 + ((lane >> 4) << 4);
    __bf16* o = out + ((size_t)tile * 32 + lane) * 16;
#pragma unroll 4
    for (int i = 0; i < 16; ++i)
        o[i] = (__bf16)W[(size_t)(kbase + i) * N + nn];
}

// ---------------------------------------------------------------------------
// C[M x N] = A[M x K] (fp32, row-major) @ Bp (pre-packed bf16) [+ bias]
// Each wave owns a 16 x (16*NT) output strip (NT compile-time, no guards).
// Caller guarantees (blockDim/32) * NT == N/16, so every wave is fully busy
// and EXEC stays all-1s around every WMMA.
// Per 32-wide K-step: ONE A-fragment load (16 fp32 -> bf16 in regs), then NT
// back-to-back v_wmma_f32_16x16x32_bf16 against NT packed B fragments.
// A-fragment (contiguous k, same as Bp): lanes 0-15: M=lane, K = 0..15;
//                              lanes 16-31: M=lane-16, K = 16..31.
// ---------------------------------------------------------------------------
template <int NT>
__global__ __launch_bounds__(128)
void gemm_bf16_wmma(const float* __restrict__ A, const __bf16* __restrict__ Bp,
                    const float* __restrict__ bias, float* __restrict__ C,
                    int N, int K) {
    const int lane   = threadIdx.x & 31;
    const int wave   = threadIdx.x >> 5;
    const int ntiles = N >> 4;
    const int nt0    = wave * NT;
    const int mt     = blockIdx.x;
    const int ktiles = K >> 5;
    const int m      = mt * 16 + (lane & 15);
    const int kA     = (lane >> 4) << 4;            // 0 or 16
    const float* __restrict__ Arow = A + (size_t)m * K;

    v8f acc[NT];
#pragma unroll
    for (int i = 0; i < NT; ++i)
        acc[i] = (v8f){0.f, 0.f, 0.f, 0.f, 0.f, 0.f, 0.f, 0.f};

    for (int kt = 0; kt < ktiles; ++kt) {
        // ---- A fragment: 16 fp32 -> 16 bf16 in registers ----
        const float4 f0 = *(const float4*)(Arow + kt * 32 + kA);
        const float4 f1 = *(const float4*)(Arow + kt * 32 + kA + 4);
        const float4 f2 = *(const float4*)(Arow + kt * 32 + kA + 8);
        const float4 f3 = *(const float4*)(Arow + kt * 32 + kA + 12);
        v16bf a;
        a[0]  = (__bf16)f0.x; a[1]  = (__bf16)f0.y; a[2]  = (__bf16)f0.z; a[3]  = (__bf16)f0.w;
        a[4]  = (__bf16)f1.x; a[5]  = (__bf16)f1.y; a[6]  = (__bf16)f1.z; a[7]  = (__bf16)f1.w;
        a[8]  = (__bf16)f2.x; a[9]  = (__bf16)f2.y; a[10] = (__bf16)f2.z; a[11] = (__bf16)f2.w;
        a[12] = (__bf16)f3.x; a[13] = (__bf16)f3.y; a[14] = (__bf16)f3.z; a[15] = (__bf16)f3.w;

        // ---- NT WMMAs re-using the A fragment ----
        const __bf16* Bbase = Bp + ((size_t)(kt * ntiles + nt0) * 32 + lane) * 16;
#pragma unroll
        for (int i = 0; i < NT; ++i) {
            const v16bf b = *(const v16bf*)(Bbase + (size_t)i * 512);  // next ntile frag
            acc[i] = __builtin_amdgcn_wmma_f32_16x16x32_bf16(
                         false, a, false, b, (short)0, acc[i], false, false);
        }
    }

    // 32-bit C/D 16x16: lanes 0-15: N=lane, M=vgpr; lanes 16-31: N=lane-16, M=8+vgpr
    const int mbase = mt * 16 + ((lane >> 4) << 3);
#pragma unroll
    for (int i = 0; i < NT; ++i) {
        const int nn = (nt0 + i) * 16 + (lane & 15);
        const float bv = bias ? bias[nn] : 0.0f;
#pragma unroll
        for (int r = 0; r < 8; ++r)
            C[(size_t)(mbase + r) * N + nn] = acc[i][r] + bv;
    }
}

// ---------------------------------------------------------------------------
// Edge kernels. One wave per edge; lane covers feature j*32+lane, j=0..7.
// Virtual self-loops: eid >= E  ->  src = dst = eid - E.
// ---------------------------------------------------------------------------
__device__ __forceinline__ void edge_endpoints(const long long* ei, int E, long long eid,
                                               int& src, int& dst) {
    if (eid < E) { src = (int)ei[eid]; dst = (int)ei[(size_t)E + eid]; }
    else         { src = dst = (int)(eid - E); }
}

__global__ __launch_bounds__(256)
void gat_scores(const long long* __restrict__ ei, int E, int n,
                const float* __restrict__ xl, const float* __restrict__ xr,
                const float* __restrict__ att,
                float* __restrict__ p, unsigned* __restrict__ emax_enc) {
    const int lane = threadIdx.x & 31;
    const int wave = threadIdx.x >> 5;
    const long long eid = (long long)blockIdx.x * 8 + wave;
    const long long E_tot = (long long)E + n;
    if (eid >= E_tot) return;
    int src, dst;
    edge_endpoints(ei, E, eid, src, dst);
    const float* pl = xl + (size_t)src * HH;
    const float* pr = xr + (size_t)dst * HH;
    float acc0 = 0.f, acc1 = 0.f, acc2 = 0.f, acc3 = 0.f;
#pragma unroll
    for (int j = 0; j < 8; ++j) {
        const int idx = j * 32 + lane;
        float s = pl[idx] + pr[idx];
        s = s > 0.f ? s : 0.2f * s;              // LeakyReLU(0.2)
        const float w = s * att[idx];            // att[h*64 + d] == att[idx]
        if (j < 2) acc0 += w; else if (j < 4) acc1 += w;
        else if (j < 6) acc2 += w; else acc3 += w;
    }
#pragma unroll
    for (int m = 16; m >= 1; m >>= 1) {
        acc0 += __shfl_xor(acc0, m, 32);
        acc1 += __shfl_xor(acc1, m, 32);
        acc2 += __shfl_xor(acc2, m, 32);
        acc3 += __shfl_xor(acc3, m, 32);
    }
    if (lane == 0) {
        *(float4*)(p + (size_t)eid * 4) = make_float4(acc0, acc1, acc2, acc3);
        float ev[4] = {acc0, acc1, acc2, acc3};
#pragma unroll
        for (int h = 0; h < 4; ++h) {
            unsigned u = __float_as_uint(ev[h]);
            u = (u & 0x80000000u) ? ~u : (u | 0x80000000u);  // order-preserving encode
            atomicMax(&emax_enc[(size_t)dst * 4 + h], u);
        }
    }
}

__global__ __launch_bounds__(256)
void gat_exp(const long long* __restrict__ ei, int E, int n,
             const unsigned* __restrict__ emax_enc,
             float* __restrict__ p, float* __restrict__ den) {
    const long long t = (long long)blockIdx.x * blockDim.x + threadIdx.x;
    const long long tot = ((long long)E + n) * 4;
    if (t >= tot) return;
    const long long eid = t >> 2;
    const int h = (int)(t & 3);
    int src, dst;
    edge_endpoints(ei, E, eid, src, dst);
    const unsigned u = emax_enc[(size_t)dst * 4 + h];
    const unsigned bits = (u & 0x80000000u) ? (u ^ 0x80000000u) : ~u;  // decode
    const float emax = __uint_as_float(bits);
    const float pe = expf(p[t] - emax);
    p[t] = pe;
    atomicAdd(&den[(size_t)dst * 4 + h], pe);
}

__global__ __launch_bounds__(256)
void gat_aggregate(const long long* __restrict__ ei, int E, int n,
                   const float* __restrict__ xl, const float* __restrict__ p,
                   const float* __restrict__ den, float* __restrict__ out) {
    const int lane = threadIdx.x & 31;
    const int wave = threadIdx.x >> 5;
    const long long eid = (long long)blockIdx.x * 8 + wave;
    const long long E_tot = (long long)E + n;
    if (eid >= E_tot) return;
    int src, dst;
    edge_endpoints(ei, E, eid, src, dst);
    const float4 pv = *(const float4*)(p + (size_t)eid * 4);
    const float* dn = den + (size_t)dst * 4;
    const float al[4] = {pv.x / (dn[0] + 1e-16f), pv.y / (dn[1] + 1e-16f),
                         pv.z / (dn[2] + 1e-16f), pv.w / (dn[3] + 1e-16f)};
    const float* pl = xl + (size_t)src * HH;
    float* po = out + (size_t)dst * HH;
#pragma unroll
    for (int j = 0; j < 8; ++j) {
        const int idx = j * 32 + lane;
        atomicAdd(&po[idx], al[j >> 1] * pl[idx]);
    }
}

__global__ __launch_bounds__(256)
void finalize_layer(const float* __restrict__ acc, const float* __restrict__ bias,
                    float* __restrict__ out, long long total, int do_elu) {
    const long long i = (long long)blockIdx.x * blockDim.x + threadIdx.x;
    if (i >= total) return;
    float f = acc[i] + bias[i & (HH - 1)];
    if (do_elu) f = f > 0.f ? f : expm1f(f);
    out[i] = f;
}

__global__ __launch_bounds__(256)
void heads_kernel(const float* __restrict__ h,
                  const float* __restrict__ Wd, const float* __restrict__ bd,
                  const float* __restrict__ Wu, const float* __restrict__ bu,
                  float* __restrict__ demand, float* __restrict__ urgency, int n) {
    const int lane = threadIdx.x & 31;
    const int wave = threadIdx.x >> 5;
    const int node = blockIdx.x * 8 + wave;
    if (node >= n) return;
    const float* hp = h + (size_t)node * HH;
    float sd = 0.f, su = 0.f;
#pragma unroll
    for (int j = 0; j < 8; ++j) {
        const int idx = j * 32 + lane;
        const float v = hp[idx];
        sd += v * Wd[idx];
        su += v * Wu[idx];
    }
#pragma unroll
    for (int m = 16; m >= 1; m >>= 1) {
        sd += __shfl_xor(sd, m, 32);
        su += __shfl_xor(su, m, 32);
    }
    if (lane == 0) {
        demand[node]  = sd + bd[0];
        urgency[node] = su + bu[0];
    }
}

// ---------------------------------------------------------------------------
extern "C" void kernel_launch(void* const* d_in, const int* in_sizes, int n_in,
                              void* d_out, int out_size, void* d_ws, size_t ws_size,
                              hipStream_t stream) {
    const float*     x   = (const float*)d_in[0];
    const long long* ei  = (const long long*)d_in[1];   // int64 [2, E]
    const float*     Wp  = (const float*)d_in[2];
    const float*     bp  = (const float*)d_in[3];
    const float* Wl[3]  = {(const float*)d_in[4],  (const float*)d_in[8],  (const float*)d_in[12]};
    const float* Wr[3]  = {(const float*)d_in[5],  (const float*)d_in[9],  (const float*)d_in[13]};
    const float* att[3] = {(const float*)d_in[6],  (const float*)d_in[10], (const float*)d_in[14]};
    const float* bb[3]  = {(const float*)d_in[7],  (const float*)d_in[11], (const float*)d_in[15]};
    const float* Wd = (const float*)d_in[16];
    const float* bd = (const float*)d_in[17];
    const float* Wu = (const float*)d_in[18];
    const float* bu = (const float*)d_in[19];

    const int E = in_sizes[1] / 2;          // 500000
    const int n = in_sizes[0] / F_IN;       // 50000 (multiple of 16)
    const int E_tot = E + n;                // + self loops
    const long long NF = (long long)n * HH; // 12.8M

    // Workspace carve-up (all offsets 32B aligned)
    float*    bufA = (float*)d_ws;                       // layer input  [n, HH]
    float*    bufB = bufA + NF;                          // xl           [n, HH]
    float*    bufC = bufB + NF;                          // xr           [n, HH]
    float*    bufD = bufC + NF;                          // accumulator  [n, HH]
    float*    pbuf = bufD + NF;                          // e / p        [E_tot, 4]
    float*    den  = pbuf + (size_t)E_tot * 4;           // denominators [n, 4]
    unsigned* emax = (unsigned*)(den + (size_t)n * 4);   // encoded max  [n, 4]
    __bf16*   packs = (__bf16*)(emax + (size_t)n * 4);   // packed weights
    const size_t PSLOT = 65536;                          // bf16 elems per matrix slot
    __bf16* pWp   = packs;
    __bf16* pWl[3] = {packs + 1 * PSLOT, packs + 3 * PSLOT, packs + 5 * PSLOT};
    __bf16* pWr[3] = {packs + 2 * PSLOT, packs + 4 * PSLOT, packs + 6 * PSLOT};

    float* h_out   = (float*)d_out;  // [n, HH]
    float* demand  = h_out + NF;     // [n]
    float* urgency = demand + n;     // [n]

    // --- Pack all weight matrices into WMMA B-fragment order (bf16) ---
    pack_wmma_b<<<1, 64, 0, stream>>>(Wp, F_IN, F_IN, pWp);
    for (int L = 0; L < 3; ++L) {
        const int K = (L == 0) ? F_IN : HH;
        const int tot = (K >> 5) * (HH >> 4) * 32;
        pack_wmma_b<<<(tot + 127) / 128, 128, 0, stream>>>(Wl[L], K, HH, pWl[L]);
        pack_wmma_b<<<(tot + 127) / 128, 128, 0, stream>>>(Wr[L], K, HH, pWr[L]);
    }

    // --- Input projection: bufA = x @ Wp + bp  (M=n, N=32, K=32) ---
    // ntiles = 2, NT=1, 2 waves (64 threads): each wave one 16x16 tile.
    gemm_bf16_wmma<1><<<dim3(n / 16), 64, 0, stream>>>(x, pWp, bp, bufA, F_IN, F_IN);

    const int edge_blocks = (E_tot + 7) / 8;
    for (int L = 0; L < 3; ++L) {
        const int K = (L == 0) ? F_IN : HH;
        // xl = h @ Wl, xr = h @ Wr: ntiles = 16, NT=4, 4 waves (128 threads).
        gemm_bf16_wmma<4><<<dim3(n / 16), 128, 0, stream>>>(bufA, pWl[L], nullptr, bufB, HH, K);
        gemm_bf16_wmma<4><<<dim3(n / 16), 128, 0, stream>>>(bufA, pWr[L], nullptr, bufC, HH, K);
        // reset segment state + accumulator
        fill_u32<<<((long long)n * 4 + 255) / 256, 256, 0, stream>>>(emax, 0u, (long long)n * 4);
        fill_f32<<<((long long)n * 4 + 255) / 256, 256, 0, stream>>>(den, 0.f, (long long)n * 4);
        fill_f32<<<(NF + 255) / 256, 256, 0, stream>>>(bufD, 0.f, NF);
        // attention scores + segment max / softmax / aggregate
        gat_scores<<<edge_blocks, 256, 0, stream>>>(ei, E, n, bufB, bufC, att[L], pbuf, emax);
        gat_exp<<<(int)(((long long)E_tot * 4 + 255) / 256), 256, 0, stream>>>(ei, E, n, emax, pbuf, den);
        gat_aggregate<<<edge_blocks, 256, 0, stream>>>(ei, E, n, bufB, pbuf, den, bufD);
        // bias (+ ELU for layers 0,1); layer 2 writes straight into d_out
        float* dst = (L == 2) ? h_out : bufA;
        finalize_layer<<<(int)((NF + 255) / 256), 256, 0, stream>>>(bufD, bb[L], dst, NF, L < 2 ? 1 : 0);
    }

    // --- demand / urgency heads ---
    heads_kernel<<<(n + 7) / 8, 256, 0, stream>>>(h_out, Wd, bd, Wu, bu, demand, urgency, n);
}
